// MultiHeadedAttention_3_18631568130125
// MI455X (gfx1250) — matrix operand order repaired and checked by reading, not compiled, not run on hardware
//
#include <hip/hip_runtime.h>
#include <hip/hip_bf16.h>

typedef __attribute__((ext_vector_type(16))) _Float16 v16h;
typedef __attribute__((ext_vector_type(8)))  float    v8f;

#define NB    16
#define NKEYS 20000
#define DDIM  256
#define NH    8
#define NTILES_PER_B (NKEYS / 16)   // 1250
#define NTILES (NB * NTILES_PER_B)  // 20000

__device__ inline v16h cvt16(float4 x0, float4 x1, float4 x2, float4 x3) {
    v16h a;
    a[0]  = (_Float16)x0.x; a[1]  = (_Float16)x0.y; a[2]  = (_Float16)x0.z; a[3]  = (_Float16)x0.w;
    a[4]  = (_Float16)x1.x; a[5]  = (_Float16)x1.y; a[6]  = (_Float16)x1.z; a[7]  = (_Float16)x1.w;
    a[8]  = (_Float16)x2.x; a[9]  = (_Float16)x2.y; a[10] = (_Float16)x2.z; a[11] = (_Float16)x2.w;
    a[12] = (_Float16)x3.x; a[13] = (_Float16)x3.y; a[14] = (_Float16)x3.z; a[15] = (_Float16)x3.w;
    return a;
}

// Kernel 1: logits[b,h,n] = sum_d key[b,n,d] * Wk[h,d]   (q_term + bias cancel in softmax)
// One wave per 16-row n-tile. C = A(16x32 f16 key) x B(32x16 f16 Wk^T), 8 k-steps over D=256.
__global__ __launch_bounds__(256) void kterm_wmma_kernel(
    const float* __restrict__ key,   // [16, 20000, 256]
    const float* __restrict__ W,     // [8, 512]; Wk[h,d] = W[h*512 + 256 + d]
    float* __restrict__ out)         // [16, 8, 20000] logits
{
    // ---- Stage Wk^T fragments (f16, WMMA B layout) into LDS: 8 k-steps x 32 lanes x 32B = 8KB
    __shared__ v16h ldsB[8 * 32];
    {
        const int t  = threadIdx.x;      // 256 threads == 8*32 fragments
        const int s  = t >> 5;           // k-step
        const int L  = t & 31;           // lane slot
        const int hh = L & 15;           // B column (head), zero-pad h>=8
        const int kb = s * 32 + ((L >> 4) & 1) * 8;  // K rows: lanes 0-15 -> {0..7,16..23}, 16-31 -> {8..15,24..31}
        v16h frag;
        if (hh < NH) {
            const float* wp = W + hh * (2 * DDIM) + DDIM + kb;
            #pragma unroll
            for (int r = 0; r < 16; ++r) frag[r] = (_Float16)wp[r + (r >> 3) * 8];
        } else {
            #pragma unroll
            for (int r = 0; r < 16; ++r) frag[r] = (_Float16)0.f;
        }
        ldsB[s * 32 + L] = frag;
    }
    __syncthreads();

    const int lane = threadIdx.x & 31;
    const int wave = threadIdx.x >> 5;
    const int tile = blockIdx.x * 8 + wave;          // 0..19999, exact
    const int b    = tile / NTILES_PER_B;
    const int n0   = (tile - b * NTILES_PER_B) * 16;

    const int m      = lane & 15;                    // A row within tile
    const int hiHalf = (lane >> 4) & 1;
    const int c0     = hiHalf * 8;                   // A K-base: lanes<16 -> {0..7,16..23}, else {8..15,24..31}

    const float* arow = key + ((size_t)b * NKEYS + (n0 + m)) * DDIM + c0;

    v8f c = {};
    #pragma unroll
    for (int s = 0; s < 8; ++s) {
        const float4* ap = reinterpret_cast<const float4*>(arow + s * 32);
        float4 x0 = ap[0];   // K = c0 .. c0+3
        float4 x1 = ap[1];   // K = c0+4 .. c0+7
        float4 x2 = ap[4];   // K = c0+16 .. c0+19
        float4 x3 = ap[5];   // K = c0+20 .. c0+23
        v16h a  = cvt16(x0, x1, x2, x3);
        v16h bf = ldsB[s * 32 + lane];
        c = __builtin_amdgcn_wmma_f32_16x16x32_f16(
                /*neg_a=*/false, a, /*neg_b=*/false, bf,
                /*c_mod=*/(short)0, c, /*reuse_a=*/false, /*reuse_b=*/false);
    }

    // C layout: vgpr r, lane L -> (M = r + 8*hiHalf, N = L&15). 8 consecutive n per lane.
    const int h = lane & 15;
    if (h < NH) {
        float* op = out + ((size_t)(b * NH + h)) * NKEYS + n0 + hiHalf * 8;
        *reinterpret_cast<float4*>(op)     = make_float4(c[0], c[1], c[2], c[3]);
        *reinterpret_cast<float4*>(op + 4) = make_float4(c[4], c[5], c[6], c[7]);
    }
}

// Kernel 2: in-place softmax over n for each (b,h) row. One block per row; deterministic
// fixed-partition loops + shared-memory tree reductions.
__global__ __launch_bounds__(256) void softmax_rows_kernel(float* __restrict__ out)
{
    __shared__ float red[256];
    float* p = out + (size_t)blockIdx.x * NKEYS;     // blockIdx.x = b*8 + h, 128 blocks
    const int tid = threadIdx.x;

    // pass 1: max
    float m = -3.402823466e38f;
    for (int i = tid; i < NKEYS; i += 256) m = fmaxf(m, p[i]);
    red[tid] = m;
    __syncthreads();
    #pragma unroll
    for (int off = 128; off > 0; off >>= 1) {
        if (tid < off) red[tid] = fmaxf(red[tid], red[tid + off]);
        __syncthreads();
    }
    const float M = red[0];
    __syncthreads();

    // pass 2: exp (stored in place) + sum
    float s = 0.f;
    for (int i = tid; i < NKEYS; i += 256) {
        float e = __expf(p[i] - M);
        p[i] = e;
        s += e;
    }
    red[tid] = s;
    __syncthreads();
    #pragma unroll
    for (int off = 128; off > 0; off >>= 1) {
        if (tid < off) red[tid] += red[tid + off];
        __syncthreads();
    }
    const float inv = 1.0f / red[0];
    __syncthreads();

    // pass 3: normalize
    for (int i = tid; i < NKEYS; i += 256) p[i] *= inv;
}

extern "C" void kernel_launch(void* const* d_in, const int* in_sizes, int n_in,
                              void* d_out, int out_size, void* d_ws, size_t ws_size,
                              hipStream_t stream) {
    // inputs (setup_inputs order): query [16,1,256] (cancels in softmax, unused),
    // key [16,20000,256], W [8,512], b [8] (cancels, unused)
    const float* key = (const float*)d_in[1];
    const float* W   = (const float*)d_in[2];
    float* out       = (float*)d_out;                // [16, 8, 20000]

    kterm_wmma_kernel<<<NTILES / 8, 256, 0, stream>>>(key, W, out);
    softmax_rows_kernel<<<NB * NH, 256, 0, stream>>>(out);
}
